// GNNConditionAttn_56349970923587
// MI455X (gfx1250) — matrix operand order repaired and checked by reading, not compiled, not run on hardware
//
#include <hip/hip_runtime.h>
#include <hip/hip_bf16.h>
#include <stdint.h>

// Problem sizes (match reference)
#define BB 16
#define NN 128
#define DD 128
#define HH 8
#define HDD 16
#define LL 2
#define CC0 64
#define CC1 64

typedef __attribute__((ext_vector_type(16))) __bf16 v16bf;
typedef __attribute__((ext_vector_type(8)))  float  v8f;
typedef unsigned int u32x4 __attribute__((ext_vector_type(4)));
typedef int          i32x8 __attribute__((ext_vector_type(8)));
typedef int          i32x4 __attribute__((ext_vector_type(4)));

union AB16 { v16bf v; uint32_t u[8]; };
union F8   { v8f v; float f[8]; };

// Use the Tensor Data Mover for the contiguous r-tile load if available.
// This toolchain exposes the 6-arg form:
//   (u32x4 g0, i32x8 g1, i32x4 g2, i32x4 g3, i32x8 gx, i32 cpol)
#ifndef USE_TDM
#if defined(__HIP_DEVICE_COMPILE__) && defined(__has_builtin)
#if __has_builtin(__builtin_amdgcn_tensor_load_to_lds) && \
    __has_builtin(__builtin_amdgcn_s_wait_tensorcnt)
#define USE_TDM 1
#else
#define USE_TDM 0
#endif
#else
#define USE_TDM 0
#endif
#endif

__device__ __forceinline__ v8f zero8() {
  v8f z = {0.f,0.f,0.f,0.f,0.f,0.f,0.f,0.f};
  return z;
}
__device__ __forceinline__ uint16_t f32_to_bf16(float f) {
  uint32_t u = __float_as_uint(f);
  uint32_t r = (u + 0x7FFFu + ((u >> 16) & 1u)) >> 16;
  return (uint16_t)r;
}
__device__ __forceinline__ float bf2f(uint16_t h) {
  return __uint_as_float(((uint32_t)h) << 16);
}

// ---- WMMA fragment loaders (wave32, documented CDNA5 VGPR layouts) ----
// A: row-major [M][K] bf16 tile 16x32 at (m0,k0). VGPR v holds K pair:
//   v<4 : K = 2v + 8g ; v>=4 : K = 16 + 2(v-4) + 8g   (g = lane>>4)
__device__ __forceinline__ v16bf load_a_frag(const uint16_t* A, int lda,
                                             int m0, int k0, int lane) {
  int g = lane >> 4, m = lane & 15;
  const uint32_t* row = (const uint32_t*)(A + (size_t)(m0 + m) * lda + k0);
  AB16 f;
#pragma unroll
  for (int v = 0; v < 8; ++v) {
    int kp = (v < 4) ? (v + g * 4) : (v + 4 + g * 4);
    f.u[v] = row[kp];
  }
  return f.v;
}
// B stored transposed (N-major): Bt[n][k]. Fragment K order matches A (table order).
__device__ __forceinline__ v16bf load_b_frag(const uint16_t* Bt, int ldb,
                                             int n0, int k0, int lane) {
  int g = lane >> 4, n = lane & 15;
  const uint32_t* row = (const uint32_t*)(Bt + (size_t)(n0 + n) * ldb + k0);
  AB16 f;
#pragma unroll
  for (int v = 0; v < 8; ++v) f.u[v] = row[(v < 4) ? (v + g * 4) : (v + 4 + g * 4)];
  return f.v;
}
// One 16x16 output tile over K=128 (4 WMMA k-steps), A and Bt both 128 wide.
__device__ __forceinline__ v8f mm128(const uint16_t* A, const uint16_t* Bt,
                                     int m0, int n0, int lane, v8f acc) {
#pragma unroll
  for (int kt = 0; kt < 4; ++kt) {
    v16bf a = load_a_frag(A, 128, m0, kt * 32, lane);
    v16bf b = load_b_frag(Bt, 128, n0, kt * 32, lane);
    acc = __builtin_amdgcn_wmma_f32_16x16x32_bf16(false, a, false, b,
                                                  (short)0, acc, false, false);
  }
  return acc;
}
// C/D lane coords: row = m0 + v + 8*(lane>>4), col = n0 + (lane&15)
__device__ __forceinline__ void tile_rc(int lane, int v, int m0, int n0,
                                        int& r, int& c) {
  r = m0 + v + ((lane >> 4) << 3);
  c = n0 + (lane & 15);
}
// Load a 128x128 f32 weight block transposed to bf16 LDS: wt[n*128+k] = W[k][coff+n]
__device__ __forceinline__ void load_wt(uint16_t* wt, const float* __restrict__ W,
                                        int ldw, int coff, int tid) {
  for (int i = tid; i < 128 * 128; i += 256) {
    int k = i >> 7, n = i & 127;
    wt[n * 128 + k] = f32_to_bf16(W[(size_t)k * ldw + coff + n]);
  }
}
// In-place LayerNorm on a 128-wide bf16 LDS row (one thread per row)
__device__ __forceinline__ void ln_row_bf16(uint16_t* row,
                                            const float* __restrict__ g,
                                            const float* __restrict__ b) {
  float mu = 0.f;
  for (int d = 0; d < 128; ++d) mu += bf2f(row[d]);
  mu *= (1.f / 128.f);
  float var = 0.f;
  for (int d = 0; d < 128; ++d) { float t = bf2f(row[d]) - mu; var += t * t; }
  var *= (1.f / 128.f);
  float rs = rsqrtf(var + 1e-5f);
  for (int d = 0; d < 128; ++d)
    row[d] = f32_to_bf16((bf2f(row[d]) - mu) * rs * g[d] + b[d]);
}

// =================== sparse edge index maps ===================
__global__ void k_init_maps(int* __restrict__ map0, int* __restrict__ mapF,
                            int* __restrict__ mapB) {
  int i = blockIdx.x * 256 + threadIdx.x;
  if (i < BB * NN) map0[i] = -1;
  if (i < BB * NN * NN) { mapF[i] = -1; mapB[i] = -1; }
}

__global__ void k_scatter(const uint8_t* __restrict__ mask0,
                          const int* __restrict__ pidx0,
                          const uint8_t* __restrict__ mask1,
                          const int* __restrict__ pidx1,
                          int* __restrict__ map0, int* __restrict__ mapF,
                          int* __restrict__ mapB) {
  int i = blockIdx.x * 256 + threadIdx.x;
  if (i >= BB * 64) return;
  int b = i >> 6, c = i & 63;
  if (mask0[i]) {
    int s = pidx0[i];
    if ((unsigned)s < (unsigned)NN) map0[b * NN + s] = c;
  }
  if (mask1[i]) {
    int s = pidx1[2 * i + 0], t = pidx1[2 * i + 1];
    if ((unsigned)s < (unsigned)NN && (unsigned)t < (unsigned)NN) {
      mapF[((size_t)b * NN + s) * NN + t] = c;  // emd1[:, :D] at (s,t)
      mapB[((size_t)b * NN + t) * NN + s] = c;  // emd1[:, D:] at (t,s)
    }
  }
}

// =================== r = eattr + fourier_pe, stored bf16 as rT[b][t][s][d] ===
__global__ __launch_bounds__(128) void k_build_r(
    const float* __restrict__ position, const float* __restrict__ heading,
    const float* __restrict__ emd0, const float* __restrict__ emd1,
    const int* __restrict__ map0, const int* __restrict__ mapF,
    const int* __restrict__ mapB, uint16_t* __restrict__ rT,
    uint8_t* __restrict__ emaskT) {
  int bid = blockIdx.x;
  int s = bid & 127, t = (bid >> 7) & 127, b = bid >> 14;
  int d = threadIdx.x;

  float pxs = position[(size_t)(b * NN + s) * 2 + 0];
  float pys = position[(size_t)(b * NN + s) * 2 + 1];
  float pxt = position[(size_t)(b * NN + t) * 2 + 0];
  float pyt = position[(size_t)(b * NN + t) * 2 + 1];
  float hs = heading[b * NN + s], ht = heading[b * NN + t];
  float rx = pxs - pxt, ry = pys - pyt;          // rel_pos[b,s,t]
  float dist = sqrtf(rx * rx + ry * ry);
  const float PI = 3.14159265358979323846f;
  float a = hs - ht + PI;
  float m = fmodf(a, 2.f * PI);
  if (m < 0.f) m += 2.f * PI;
  float rel_ori = m - PI;
  float ct = __cosf(ht), st = __sinf(ht);
  float cross = ct * ry - st * rx;
  float dotv  = ct * rx + st * ry;
  float rel_ori_vec = atan2f(cross, dotv);
  float pev[4] = {dist, rel_ori, rel_ori_vec, rel_ori_vec};

  int comp = d >> 5, j = d & 31;
  float freq = __powf(10000.f, (float)(2 * (j >> 1)) * (1.f / 32.f));
  float ph = pev[comp] / freq;
  float pe = (j & 1) ? __cosf(ph) : __sinf(ph);

  int m0i = map0[b * NN + s];
  bool m0 = (s == t) && (m0i >= 0);
  int fi = mapF[((size_t)b * NN + s) * NN + t];
  int bi = mapB[((size_t)b * NN + s) * NN + t];
  bool m1 = (fi >= 0) || (bi >= 0);
  float cnt = fmaxf((float)((m0 ? 1 : 0) + (m1 ? 1 : 0)), 1.f);
  float ea = 0.f;
  if (m0) ea += emd0[((size_t)(b * CC0 + m0i)) * DD + d];
  if (bi >= 0)      ea += emd1[((size_t)(b * CC1 + bi)) * (2 * DD) + DD + d];
  else if (fi >= 0) ea += emd1[((size_t)(b * CC1 + fi)) * (2 * DD) + d];

  float rv = ea / cnt + pe;
  rT[(((size_t)(b * NN + t)) * NN + s) * DD + d] = f32_to_bf16(rv);
  if (d == 0)
    emaskT[((size_t)(b * NN + t)) * NN + s] = (m0 || m1) ? 1 : 0;
}

// =================== elementwise helpers ===================
__global__ void k_copy(const float* __restrict__ src, float* __restrict__ dst,
                       int n) {
  int i = blockIdx.x * 256 + threadIdx.x;
  if (i < n) dst[i] = src[i];
}
__global__ void k_final(const float* __restrict__ pe,
                        const uint8_t* __restrict__ pm,
                        const float* __restrict__ x, float* __restrict__ out,
                        int n) {
  int i = blockIdx.x * 256 + threadIdx.x;
  if (i < n) {
    int tok = i >> 7;
    out[i] = pm[tok] ? pe[i] + x[i] : pe[i];
  }
}

// =================== per-layer: LN(x) + QKV projections (WMMA) ============
__global__ __launch_bounds__(256) void k_qkv(
    const float* __restrict__ xws, float* __restrict__ xnws,
    float* __restrict__ qws, float* __restrict__ kws, float* __restrict__ vws,
    const float* __restrict__ Wq, const float* __restrict__ bq,
    const float* __restrict__ Wk, const float* __restrict__ Wv,
    const float* __restrict__ bv, const float* __restrict__ lng,
    const float* __restrict__ lnb) {
  __shared__ __align__(16) uint8_t smem[131072];
  float*    xf32 = (float*)smem;                  // 64K
  uint16_t* xnb  = (uint16_t*)(smem + 65536);     // 32K
  uint16_t* wt   = (uint16_t*)(smem + 98304);     // 32K

  int tid = threadIdx.x, lane = tid & 31, w = tid >> 5, m0 = w * 16;
  size_t base = (size_t)blockIdx.x * 128 * 128;

  for (int i = tid; i < 128 * 128; i += 256) xf32[i] = xws[base + i];
  // warm L2 with the other two weight matrices while we LN + GEMM Wq
  __builtin_prefetch(Wk + (tid << 6), 0, 1);
  __builtin_prefetch(Wv + (tid << 6), 0, 1);
  __syncthreads();
  if (tid < 128) {
    float* row = xf32 + tid * 128;
    float mu = 0.f;
    for (int d = 0; d < 128; ++d) mu += row[d];
    mu *= (1.f / 128.f);
    float var = 0.f;
    for (int d = 0; d < 128; ++d) { float tv = row[d] - mu; var += tv * tv; }
    float rs = rsqrtf(var * (1.f / 128.f) + 1e-5f);
    for (int d = 0; d < 128; ++d) {
      float xv = (row[d] - mu) * rs * lng[d] + lnb[d];
      xnws[base + tid * 128 + d] = xv;
      xnb[tid * 128 + d] = f32_to_bf16(xv);
    }
  }
  __syncthreads();

  // ---- Q = xn @ Wq + bq ----
  load_wt(wt, Wq, 128, 0, tid);
  __syncthreads();
  for (int nT = 0; nT < 8; ++nT) {
    F8 u; u.v = mm128(xnb, wt, m0, nT * 16, lane, zero8());
#pragma unroll
    for (int v = 0; v < 8; ++v) {
      int r, c; tile_rc(lane, v, m0, nT * 16, r, c);
      qws[base + (size_t)r * 128 + c] = u.f[v] + bq[c];
    }
  }
  __syncthreads();
  // ---- K = xn @ Wk ----
  load_wt(wt, Wk, 128, 0, tid);
  __syncthreads();
  for (int nT = 0; nT < 8; ++nT) {
    F8 u; u.v = mm128(xnb, wt, m0, nT * 16, lane, zero8());
#pragma unroll
    for (int v = 0; v < 8; ++v) {
      int r, c; tile_rc(lane, v, m0, nT * 16, r, c);
      kws[base + (size_t)r * 128 + c] = u.f[v];
    }
  }
  __syncthreads();
  // ---- V = xn @ Wv + bv ----
  load_wt(wt, Wv, 128, 0, tid);
  __syncthreads();
  for (int nT = 0; nT < 8; ++nT) {
    F8 u; u.v = mm128(xnb, wt, m0, nT * 16, lane, zero8());
#pragma unroll
    for (int v = 0; v < 8; ++v) {
      int r, c; tile_rc(lane, v, m0, nT * 16, r, c);
      vws[base + (size_t)r * 128 + c] = u.f[v] + bv[c];
    }
  }
}

// =================== per-layer: simk[b][t][s][h] = q[t,h]·k[s,h] ==========
__global__ __launch_bounds__(256) void k_simk(const float* __restrict__ qws,
                                              const float* __restrict__ kws,
                                              float* __restrict__ simk) {
  __shared__ float qrow[128];
  int bid = blockIdx.x;
  int t = bid & 127, b = bid >> 7;
  int tid = threadIdx.x;
  if (tid < 128) qrow[tid] = qws[((size_t)(b * NN + t)) * DD + tid];
  __syncthreads();
  for (int p = tid; p < NN * HH; p += 256) {
    int s = p >> 3, h = p & 7;
    const float* kr = kws + ((size_t)(b * NN + s)) * DD + h * HDD;
    float acc = 0.f;
#pragma unroll
    for (int d = 0; d < HDD; ++d) acc += qrow[h * HDD + d] * kr[d];
    simk[(((size_t)(b * NN + t)) * NN + s) * HH + h] = acc;
  }
}

// =================== per-layer fused relational attention ==================
// Block = one (b,t). rn = LN(r tile); KR = rn@Wkr fused into sim via lane
// reductions (never stored); VR = rn@Wvr + bvr kept bf16 in LDS; softmax; agg.
__global__ __launch_bounds__(256) void k_attn(
    const uint16_t* __restrict__ rT, const uint8_t* __restrict__ emaskT,
    const float* __restrict__ simk, const float* __restrict__ qws,
    const float* __restrict__ vws, const float* __restrict__ Wkr,
    const float* __restrict__ Wvr, const float* __restrict__ bvr,
    const float* __restrict__ lng, const float* __restrict__ lnb,
    float* __restrict__ aggws) {
  __shared__ __align__(16) uint16_t rbuf[128 * 128];
  __shared__ __align__(16) uint16_t wt[128 * 128];
  __shared__ __align__(16) uint16_t vrb[128 * 128];
  __shared__ __align__(16) uint16_t vbuf[128 * 128];
  __shared__ float simbuf[128 * 8];
  __shared__ float qbuf[128];

  int bid = blockIdx.x;
  int t = bid & 127, b = bid >> 7;
  int tid = threadIdx.x, lane = tid & 31, w = tid >> 5, m0 = w * 16;

  // contiguous 32KB tile: rT[b][t][*][*]
  const uint16_t* rg = rT + (((size_t)(b * NN + t)) * NN) * DD;
#if USE_TDM
  // Tensor Data Mover: one DMA descriptor moves the whole 32KB tile to LDS.
  // D# group0: count=1 | lds_addr | global_addr | type=2 ("image").
  // D# group1: data_size=2B, 1-D tile: tensor_dim0 = tile_dim0 = 16384 elems.
  if (w == 0) {
    uint64_t ga = (uint64_t)(uintptr_t)rg;
    uint32_t la = (uint32_t)(uintptr_t)(&rbuf[0]);  // flat->LDS: low 32 bits
    union { u32x4 v; unsigned int e[4]; } g0;
    g0.e[0] = 1u;                                   // count=1, user mode
    g0.e[1] = la;                                   // lds_addr
    g0.e[2] = (uint32_t)ga;                         // global_addr[31:0]
    g0.e[3] = (uint32_t)(ga >> 32) | (2u << 30);    // ga[56:32] | type=2
    union { i32x8 v; int e[8]; } g1;
    g1.e[0] = 0x00010000;          // wg_mask=0, data_size=1 (2 bytes)
    g1.e[1] = 0x40000000;          // tensor_dim0[15:0]=16384 at bits[63:48]
    g1.e[2] = 0x00010000;          // tensor_dim1=1
    g1.e[3] = 0x40000000;          // tile_dim0=16384 at bits[127:112]
    g1.e[4] = 0;                   // tile_dim1=0, tile_dim2=0
    g1.e[5] = 16384;               // tensor_dim0_stride[31:0]
    g1.e[6] = 0;
    g1.e[7] = 0;
    union { i32x4 v; int e[4]; } gz4;
    gz4.e[0] = 0; gz4.e[1] = 0; gz4.e[2] = 0; gz4.e[3] = 0;
    union { i32x8 v; int e[8]; } gz8;
#pragma unroll
    for (int i = 0; i < 8; ++i) gz8.e[i] = 0;
    __builtin_amdgcn_tensor_load_to_lds(g0.v, g1.v, gz4.v, gz4.v, gz8.v, 0);
    __builtin_amdgcn_s_wait_tensorcnt(0);
  }
#else
  {
    const uint32_t* rg32 = (const uint32_t*)rg;
    uint32_t* rb32 = (uint32_t*)rbuf;
    for (int i = tid; i < 128 * 64; i += 256) rb32[i] = rg32[i];
  }
#endif
  const float* vg = vws + (size_t)b * NN * DD;
  for (int i = tid; i < 128 * 128; i += 256) vbuf[i] = f32_to_bf16(vg[i]);
  if (tid < 128) qbuf[tid] = qws[((size_t)(b * NN + t)) * DD + tid];
  __syncthreads();

  if (tid < 128) ln_row_bf16(rbuf + tid * 128, lng, lnb);
  load_wt(wt, Wkr, 128, 0, tid);
  __builtin_prefetch(Wvr + (tid << 6), 0, 1);
  __syncthreads();

  // GEMM1: KR strip, fused q-dot reduction (h == nT, HD == tile width)
  for (int nT = 0; nT < 8; ++nT) {
    F8 u; u.v = mm128(rbuf, wt, m0, nT * 16, lane, zero8());
    int g = lane >> 4, n = lane & 15;
    float qv = qbuf[nT * 16 + n];
#pragma unroll
    for (int v = 0; v < 8; ++v) {
      float p = u.f[v] * qv;
      p += __shfl_xor(p, 8);
      p += __shfl_xor(p, 4);
      p += __shfl_xor(p, 2);
      p += __shfl_xor(p, 1);
      if (n == 0) simbuf[(m0 + v + g * 8) * 8 + nT] = p;
    }
  }
  __syncthreads();

  load_wt(wt, Wvr, 128, 0, tid);
  __syncthreads();
  // GEMM2: VR = rn@Wvr + bvr -> LDS bf16
  for (int nT = 0; nT < 8; ++nT) {
    F8 u; u.v = mm128(rbuf, wt, m0, nT * 16, lane, zero8());
#pragma unroll
    for (int v = 0; v < 8; ++v) {
      int r, c; tile_rc(lane, v, m0, nT * 16, r, c);
      vrb[r * 128 + c] = f32_to_bf16(u.f[v] + bvr[c]);
    }
  }
  __syncthreads();

  // softmax over s per head (wave w owns head h=w)
  {
    int h = w;
    const float* sk = simk + (((size_t)(b * NN + t)) * NN) * HH;
    const uint8_t* mrow = emaskT + ((size_t)(b * NN + t)) * NN;
    float sv[4], mv[4];
    float mx = -3.0e38f;
#pragma unroll
    for (int i = 0; i < 4; ++i) {
      int s = lane + 32 * i;
      float mk = mrow[s] ? 1.f : 0.f;
      float sim = (sk[s * 8 + h] + simbuf[s * 8 + h]) * 0.25f;
      sim = (mk > 0.f) ? sim : -1.0e30f;
      sv[i] = sim; mv[i] = mk;
      mx = fmaxf(mx, sim);
    }
    for (int o = 16; o >= 1; o >>= 1) mx = fmaxf(mx, __shfl_xor(mx, o));
    float sum = 0.f;
#pragma unroll
    for (int i = 0; i < 4; ++i) { sv[i] = __expf(sv[i] - mx) * mv[i]; sum += sv[i]; }
    for (int o = 16; o >= 1; o >>= 1) sum += __shfl_xor(sum, o);
    float inv = 1.f / fmaxf(sum, 1e-20f);
#pragma unroll
    for (int i = 0; i < 4; ++i) simbuf[(lane + 32 * i) * 8 + h] = sv[i] * inv;
  }
  __syncthreads();

  // agg[d] = sum_s alpha[s,h]*(v[s,d] + vr[s,d])
  if (tid < 128) {
    int d = tid, h = d >> 4;
    float a = 0.f;
    for (int s = 0; s < 128; ++s)
      a += simbuf[s * 8 + h] * (bf2f(vbuf[s * 128 + d]) + bf2f(vrb[s * 128 + d]));
    aggws[((size_t)(b * NN + t)) * DD + d] = a;
  }
}

// =================== per-layer gating epilogue (WMMA) =====================
__global__ __launch_bounds__(256) void k_epilogue(
    const float* __restrict__ aggws, const float* __restrict__ xnws,
    float* __restrict__ xws, const float* __restrict__ Ws,
    const float* __restrict__ bs, const float* __restrict__ Wg,
    const float* __restrict__ bg, const float* __restrict__ Wo,
    const float* __restrict__ bo) {
  __shared__ __align__(16) uint16_t aggb[128 * 128];
  __shared__ __align__(16) uint16_t xnb[128 * 128];
  __shared__ __align__(16) uint16_t ob[128 * 128];
  __shared__ __align__(16) uint16_t wt[128 * 128];
  int tid = threadIdx.x, lane = tid & 31, w = tid >> 5, m0 = w * 16;
  size_t base = (size_t)blockIdx.x * 128 * 128;

  for (int i = tid; i < 128 * 128; i += 256) {
    aggb[i] = f32_to_bf16(aggws[base + i]);
    xnb[i]  = f32_to_bf16(xnws[base + i]);
  }
  __syncthreads();

  v8f S[8], G[8];
  // S = xn @ Ws
  load_wt(wt, Ws, 128, 0, tid);
  __syncthreads();
#pragma unroll
  for (int nT = 0; nT < 8; ++nT) S[nT] = mm128(xnb, wt, m0, nT * 16, lane, zero8());
  __syncthreads();
  // G = agg @ Wg[:128]
  load_wt(wt, Wg, 128, 0, tid);
  __syncthreads();
#pragma unroll
  for (int nT = 0; nT < 8; ++nT) G[nT] = mm128(aggb, wt, m0, nT * 16, lane, zero8());
  __syncthreads();
  // G += xn @ Wg[128:]
  load_wt(wt, Wg + 128 * 128, 128, 0, tid);
  __syncthreads();
#pragma unroll
  for (int nT = 0; nT < 8; ++nT) G[nT] = mm128(xnb, wt, m0, nT * 16, lane, G[nT]);
  __syncthreads();
  // out = agg + sigmoid(G+bg) * (S+bs - agg)
#pragma unroll
  for (int nT = 0; nT < 8; ++nT) {
    F8 us, ug; us.v = S[nT]; ug.v = G[nT];
#pragma unroll
    for (int v = 0; v < 8; ++v) {
      int r, c; tile_rc(lane, v, m0, nT * 16, r, c);
      float aggv = bf2f(aggb[r * 128 + c]);
      float sv = us.f[v] + bs[c];
      float gv = 1.f / (1.f + __expf(-(ug.f[v] + bg[c])));
      ob[r * 128 + c] = f32_to_bf16(aggv + gv * (sv - aggv));
    }
  }
  __syncthreads();
  // x += out @ Wo + bo
  load_wt(wt, Wo, 128, 0, tid);
  __syncthreads();
  for (int nT = 0; nT < 8; ++nT) {
    F8 u; u.v = mm128(ob, wt, m0, nT * 16, lane, zero8());
#pragma unroll
    for (int v = 0; v < 8; ++v) {
      int r, c; tile_rc(lane, v, m0, nT * 16, r, c);
      size_t idx = base + (size_t)r * 128 + c;
      xws[idx] = xws[idx] + u.f[v] + bo[c];
    }
  }
}

// =================== per-layer feed-forward (K looped over 512 hidden) =====
__global__ __launch_bounds__(256) void k_ff(
    float* __restrict__ xws, const float* __restrict__ lng,
    const float* __restrict__ lnb, const float* __restrict__ W1,
    const float* __restrict__ b1, const float* __restrict__ W2,
    const float* __restrict__ b2) {
  __shared__ __align__(16) uint8_t smem[98304];
  float*    xf32 = (float*)smem;                  // [0,64K) then reused
  uint16_t* wt   = (uint16_t*)smem;               // [0,32K)
  uint16_t* h1b  = (uint16_t*)(smem + 32768);     // [32K,64K)
  uint16_t* xfb  = (uint16_t*)(smem + 65536);     // [64K,96K)

  int tid = threadIdx.x, lane = tid & 31, w = tid >> 5, m0 = w * 16;
  size_t base = (size_t)blockIdx.x * 128 * 128;

  for (int i = tid; i < 128 * 128; i += 256) xf32[i] = xws[base + i];
  __syncthreads();
  if (tid < 128) {
    float* row = xf32 + tid * 128;
    float mu = 0.f;
    for (int d = 0; d < 128; ++d) mu += row[d];
    mu *= (1.f / 128.f);
    float var = 0.f;
    for (int d = 0; d < 128; ++d) { float tv = row[d] - mu; var += tv * tv; }
    float rs = rsqrtf(var * (1.f / 128.f) + 1e-5f);
    for (int d = 0; d < 128; ++d)
      xfb[tid * 128 + d] = f32_to_bf16((row[d] - mu) * rs * lng[d] + lnb[d]);
  }
  __syncthreads();

  v8f Dl[8];
#pragma unroll
  for (int i = 0; i < 8; ++i) Dl[i] = zero8();

  for (int c = 0; c < 4; ++c) {
    load_wt(wt, W1, 512, c * 128, tid);     // Wff1 columns [c*128, +128)
    __builtin_prefetch(W2 + (size_t)c * 128 * 128 + (tid << 6), 0, 1);
    __syncthreads();
    for (int nT = 0; nT < 8; ++nT) {
      F8 u; u.v = mm128(xfb, wt, m0, nT * 16, lane, zero8());
#pragma unroll
      for (int v = 0; v < 8; ++v) {
        int r, cc; tile_rc(lane, v, m0, nT * 16, r, cc);
        float hv = u.f[v] + b1[c * 128 + cc];
        h1b[r * 128 + cc] = f32_to_bf16(fmaxf(hv, 0.f));
      }
    }
    __syncthreads();
    load_wt(wt, W2 + (size_t)c * 128 * 128, 128, 0, tid);  // Wff2 rows chunk
    __syncthreads();
#pragma unroll
    for (int nT = 0; nT < 8; ++nT)
      Dl[nT] = mm128(h1b, wt, m0, nT * 16, lane, Dl[nT]);
    __syncthreads();
  }
#pragma unroll
  for (int nT = 0; nT < 8; ++nT) {
    F8 u; u.v = Dl[nT];
#pragma unroll
    for (int v = 0; v < 8; ++v) {
      int r, cc; tile_rc(lane, v, m0, nT * 16, r, cc);
      size_t idx = base + (size_t)r * 128 + cc;
      xws[idx] = xws[idx] + u.f[v] + b2[cc];
    }
  }
}

// =================== host-side orchestration ===================
extern "C" void kernel_launch(void* const* d_in, const int* in_sizes, int n_in,
                              void* d_out, int out_size, void* d_ws,
                              size_t ws_size, hipStream_t stream) {
  (void)in_sizes; (void)n_in; (void)out_size; (void)ws_size;
  const float*   prompt_emd  = (const float*)d_in[0];
  const uint8_t* prompt_mask = (const uint8_t*)d_in[1];
  const float*   position    = (const float*)d_in[2];
  const float*   heading     = (const float*)d_in[3];
  const float*   emd0        = (const float*)d_in[4];
  const uint8_t* mask0       = (const uint8_t*)d_in[5];
  const int*     pidx0       = (const int*)d_in[6];
  const float*   emd1        = (const float*)d_in[7];
  const uint8_t* mask1       = (const uint8_t*)d_in[8];
  const int*     pidx1       = (const int*)d_in[9];
  const float*   ln_x_g = (const float*)d_in[10];
  const float*   ln_x_b = (const float*)d_in[11];
  const float*   ln_r_g = (const float*)d_in[12];
  const float*   ln_r_b = (const float*)d_in[13];
  const float*   ln_ff_g = (const float*)d_in[14];
  const float*   ln_ff_b = (const float*)d_in[15];
  const float*   Wq = (const float*)d_in[16];
  const float*   bq = (const float*)d_in[17];
  const float*   Wk = (const float*)d_in[18];
  const float*   Wv = (const float*)d_in[19];
  const float*   bv = (const float*)d_in[20];
  const float*   Wkr = (const float*)d_in[21];
  const float*   Wvr = (const float*)d_in[22];
  const float*   bvr = (const float*)d_in[23];
  const float*   Ws  = (const float*)d_in[24];
  const float*   bs  = (const float*)d_in[25];
  const float*   Wg  = (const float*)d_in[26];
  const float*   bg  = (const float*)d_in[27];
  const float*   Wo  = (const float*)d_in[28];
  const float*   bo  = (const float*)d_in[29];
  const float*   Wff1 = (const float*)d_in[30];
  const float*   bff1 = (const float*)d_in[31];
  const float*   Wff2 = (const float*)d_in[32];
  const float*   bff2 = (const float*)d_in[33];

  uint8_t* ws = (uint8_t*)d_ws;
  size_t off = 0;
  auto carve = [&](size_t bytes) -> void* {
    void* p = ws + off;
    off += (bytes + 255) & ~(size_t)255;
    return p;
  };
  const size_t TOK = (size_t)BB * NN;                 // 2048 tokens
  uint16_t* rT     = (uint16_t*)carve((size_t)BB * NN * NN * DD * 2);
  uint8_t*  emaskT = (uint8_t*) carve((size_t)BB * NN * NN);
  int*      map0   = (int*)     carve((size_t)BB * NN * 4);
  int*      mapF   = (int*)     carve((size_t)BB * NN * NN * 4);
  int*      mapB   = (int*)     carve((size_t)BB * NN * NN * 4);
  float*    xws    = (float*)   carve(TOK * DD * 4);
  float*    xnws   = (float*)   carve(TOK * DD * 4);
  float*    qws    = (float*)   carve(TOK * DD * 4);
  float*    kws    = (float*)   carve(TOK * DD * 4);
  float*    vws    = (float*)   carve(TOK * DD * 4);
  float*    aggws  = (float*)   carve(TOK * DD * 4);
  float*    simk   = (float*)   carve((size_t)BB * NN * NN * HH * 4);

  k_init_maps<<<(BB * NN * NN + 255) / 256, 256, 0, stream>>>(map0, mapF, mapB);
  k_scatter<<<(BB * 64 + 255) / 256, 256, 0, stream>>>(mask0, pidx0, mask1,
                                                       pidx1, map0, mapF, mapB);
  k_build_r<<<BB * NN * NN, 128, 0, stream>>>(position, heading, emd0, emd1,
                                              map0, mapF, mapB, rT, emaskT);
  k_copy<<<(int)((TOK * DD + 255) / 256), 256, 0, stream>>>(prompt_emd, xws,
                                                            (int)(TOK * DD));
  for (int l = 0; l < LL; ++l) {
    const float* Wq_l  = Wq  + (size_t)l * DD * DD;
    const float* Wk_l  = Wk  + (size_t)l * DD * DD;
    const float* Wv_l  = Wv  + (size_t)l * DD * DD;
    const float* Wkr_l = Wkr + (size_t)l * DD * DD;
    const float* Wvr_l = Wvr + (size_t)l * DD * DD;
    const float* Ws_l  = Ws  + (size_t)l * DD * DD;
    const float* Wg_l  = Wg  + (size_t)l * 2 * DD * DD;
    const float* Wo_l  = Wo  + (size_t)l * DD * DD;
    const float* W1_l  = Wff1 + (size_t)l * DD * 4 * DD;
    const float* W2_l  = Wff2 + (size_t)l * 4 * DD * DD;

    k_qkv<<<BB, 256, 0, stream>>>(xws, xnws, qws, kws, vws, Wq_l,
                                  bq + l * DD, Wk_l, Wv_l, bv + l * DD,
                                  ln_x_g + l * DD, ln_x_b + l * DD);
    k_simk<<<BB * NN, 256, 0, stream>>>(qws, kws, simk);
    k_attn<<<BB * NN, 256, 0, stream>>>(rT, emaskT, simk, qws, vws, Wkr_l,
                                        Wvr_l, bvr + l * DD, ln_r_g + l * DD,
                                        ln_r_b + l * DD, aggws);
    k_epilogue<<<BB, 256, 0, stream>>>(aggws, xnws, xws, Ws_l, bs + l * DD,
                                       Wg_l, bg + l * DD, Wo_l, bo + l * DD);
    k_ff<<<BB, 256, 0, stream>>>(xws, ln_ff_g + l * DD, ln_ff_b + l * DD,
                                 W1_l, bff1 + (size_t)l * 4 * DD, W2_l,
                                 bff2 + l * DD);
  }
  k_final<<<(int)((TOK * DD + 255) / 256), 256, 0, stream>>>(
      prompt_emd, prompt_mask, xws, (float*)d_out, (int)(TOK * DD));
}
